// GAT_48438641164313
// MI455X (gfx1250) — matrix operand order repaired and checked by reading, not compiled, not run on hardware
//
#include <hip/hip_runtime.h>
#include <hip/hip_bf16.h>
#include <math.h>

#define N_NODES 50000
#define N_EDGES 800000
#define DIM_IN 64
#define DIM_H 32
#define HEADS 4
#define HD 128            // HEADS * DIM_H
#define NEG_SLOPE 0.2f

typedef __attribute__((ext_vector_type(16))) _Float16 v16h;
typedef __attribute__((ext_vector_type(8)))  float    v8f;

// ---------------------------------------------------------------------------
// WMMA GEMM: C[M x Nc] = A[M x K] * B[K x Nc] + bias (broadcast over rows).
// fp32 in memory, f16 operands, fp32 accumulate.
// One wave per 16-row tile; the wave keeps ALL Nc/16 column tiles in
// registers, so the A fragment is loaded once per k-step and reused across
// column tiles. K % 32 == 0, Nc % 16 == 0, M % 16 == 0. Fully unrolled:
// emits (K/32)*(Nc/16) v_wmma_f32_16x16x32_f16 per row tile.
// ---------------------------------------------------------------------------
template <int K, int Nc>
__global__ void gat_gemm_wmma(const float* __restrict__ A,
                              const float* __restrict__ B,
                              const float* __restrict__ bias,
                              float* __restrict__ C, int M) {
    constexpr int CT = Nc / 16;                 // column tiles per row tile
    const int lane = threadIdx.x & 31;
    const int waveInBlock = threadIdx.x >> 5;
    const int wavesPerBlock = blockDim.x >> 5;
    const int rowTiles = M >> 4;
    const int ln15 = lane & 15;
    const int abase = (lane < 16) ? 0 : 8;      // A K-base for this lane half
    const int bbase = (lane < 16) ? 0 : 8;      // B K-base for this lane half

    int wave = blockIdx.x * wavesPerBlock + waveInBlock;
    const int waveStride = gridDim.x * wavesPerBlock;

    for (int tm = wave; tm < rowTiles; tm += waveStride) {
        const int m = tm * 16 + ln15;           // A row for this lane
        v8f c[CT] = {};
        #pragma unroll
        for (int k0 = 0; k0 < K; k0 += 32) {
            v16h a;
            const float* Arow = A + (size_t)m * K + k0 + abase;
            #pragma unroll
            for (int j = 0; j < 8; ++j)  a[j] = (_Float16)Arow[j];      // K=base+j
            #pragma unroll
            for (int j = 8; j < 16; ++j) a[j] = (_Float16)Arow[j + 8];  // K=base+16+(j-8)
            #pragma unroll
            for (int tc = 0; tc < CT; ++tc) {
                v16h b;
                const float* Bcol = B + (size_t)(k0 + bbase) * Nc + tc * 16 + ln15;
                #pragma unroll
                for (int j = 0; j < 16; ++j) b[j] = (_Float16)Bcol[(size_t)(j + (j & 8)) * Nc];
                c[tc] = __builtin_amdgcn_wmma_f32_16x16x32_f16(
                            false, a, false, b, (short)0, c[tc], false, false);
            }
        }
        const int rbase = tm * 16 + ((lane < 16) ? 0 : 8);
        #pragma unroll
        for (int tc = 0; tc < CT; ++tc) {
            const int n = tc * 16 + ln15;
            const float bn = bias[n];
            #pragma unroll
            for (int r = 0; r < 8; ++r)
                C[(size_t)(rbase + r) * Nc + n] = c[tc][r] + bn;
        }
    }
}

// ---------------------------------------------------------------------------
__global__ void gat_fill(float* __restrict__ p, float v, int n) {
    int i = blockIdx.x * blockDim.x + threadIdx.x;
    if (i < n) p[i] = v;
}

__device__ __forceinline__ void atomicMaxF(float* addr, float val) {
    if (val >= 0.0f) atomicMax((int*)addr, __float_as_int(val));
    else             atomicMin((unsigned int*)addr, __float_as_uint(val));
}

// ---------------------------------------------------------------------------
// Attention scores, one WAVE per edge. Lane l covers dims {l, l+32, l+64,
// l+96} (one per head) so fs/fd rows are read as coalesced 128B transactions.
// Per-head dot products reduced across the wave with shfl_xor; fused
// segment-max into SMAX[dst,h].
// ---------------------------------------------------------------------------
__global__ void gat_scores(const float* __restrict__ FS, const float* __restrict__ FD,
                           const int* __restrict__ src, const int* __restrict__ dst,
                           const float* __restrict__ attn,
                           float* __restrict__ S, float* __restrict__ SMAX) {
    const int lane = threadIdx.x & 31;
    const int e = (blockIdx.x * blockDim.x + threadIdx.x) >> 5;
    if (e >= N_EDGES) return;
    const int u = src[e], v = dst[e];
    const float* fs = FS + (size_t)u * HD;
    const float* fd = FD + (size_t)v * HD;
    float p[HEADS];
    #pragma unroll
    for (int h = 0; h < HEADS; ++h) {
        const int dim = h * DIM_H + lane;
        float x = fs[dim] + fd[dim];
        x = x > 0.0f ? x : NEG_SLOPE * x;       // leaky_relu
        p[h] = attn[dim] * x;
    }
    #pragma unroll
    for (int off = 16; off > 0; off >>= 1) {
        #pragma unroll
        for (int h = 0; h < HEADS; ++h)
            p[h] += __shfl_xor(p[h], off, 32);
    }
    if (lane < HEADS) {
        const float sv = (lane == 0) ? p[0] : (lane == 1) ? p[1]
                       : (lane == 2) ? p[2] : p[3];
        S[e * HEADS + lane] = sv;
        atomicMaxF(&SMAX[v * HEADS + lane], sv);
    }
}

// ---------------------------------------------------------------------------
// es = exp(s - smax[dst]); accumulate denom[dst,h]. Thread per (edge,head).
// ---------------------------------------------------------------------------
__global__ void gat_expden(const int* __restrict__ dst,
                           float* __restrict__ S, const float* __restrict__ SMAX,
                           float* __restrict__ DEN) {
    int idx = blockIdx.x * blockDim.x + threadIdx.x;
    if (idx >= N_EDGES * HEADS) return;
    const int e = idx >> 2, h = idx & 3;
    const int v = dst[e];
    float es = expf(S[idx] - SMAX[v * HEADS + h]);
    S[idx] = es;
    atomicAdd(&DEN[v * HEADS + h], es);
}

// ---------------------------------------------------------------------------
// out[dst,h,:] += alpha[e,h] * fs[src,h,:], one WAVE per edge; lane l handles
// dims {l, l+32, l+64, l+96} -> coalesced loads and lane-contiguous
// global_atomic_add_f32 groups.
// ---------------------------------------------------------------------------
__global__ void gat_aggregate(const float* __restrict__ FS,
                              const int* __restrict__ src, const int* __restrict__ dst,
                              const float* __restrict__ S, const float* __restrict__ DEN,
                              float* __restrict__ OUT) {
    const int lane = threadIdx.x & 31;
    const int e = (blockIdx.x * blockDim.x + threadIdx.x) >> 5;
    if (e >= N_EDGES) return;
    const int u = src[e], v = dst[e];
    const float* fs = FS + (size_t)u * HD;
    float* outp = OUT + (size_t)v * HD;
    #pragma unroll
    for (int h = 0; h < HEADS; ++h) {
        const float alpha = S[e * HEADS + h] / DEN[v * HEADS + h];
        const int dim = h * DIM_H + lane;
        atomicAdd(&outp[dim], alpha * fs[dim]);
    }
}

// ---------------------------------------------------------------------------
// H[n,d] = (optionally relu)( mean over heads of OUT[n,h,d] )
// ---------------------------------------------------------------------------
__global__ void gat_headmean(const float* __restrict__ OUT, float* __restrict__ H,
                             int do_relu) {
    int idx = blockIdx.x * blockDim.x + threadIdx.x;
    if (idx >= N_NODES * DIM_H) return;
    const int n = idx >> 5, d = idx & 31;
    const float* row = OUT + (size_t)n * HD;
    float v = 0.25f * (row[d] + row[DIM_H + d] + row[2 * DIM_H + d] + row[3 * DIM_H + d]);
    if (do_relu) v = v > 0.0f ? v : 0.0f;
    H[idx] = v;
}

// ---------------------------------------------------------------------------
// GSUM[d] = sum over nodes of mean-over-heads OUT[n,h,d]; 32 blocks (one per
// d), deterministic tree reduction.
// ---------------------------------------------------------------------------
__global__ void gat_colsum(const float* __restrict__ OUT, float* __restrict__ GSUM) {
    __shared__ float sdata[256];
    const int d = blockIdx.x;  // 0..31
    float acc = 0.0f;
    for (int n = threadIdx.x; n < N_NODES; n += blockDim.x) {
        const float* row = OUT + (size_t)n * HD;
        acc += 0.25f * (row[d] + row[DIM_H + d] + row[2 * DIM_H + d] + row[3 * DIM_H + d]);
    }
    sdata[threadIdx.x] = acc;
    __syncthreads();
    for (int s = 128; s > 0; s >>= 1) {
        if ((int)threadIdx.x < s) sdata[threadIdx.x] += sdata[threadIdx.x + s];
        __syncthreads();
    }
    if (threadIdx.x == 0) GSUM[d] = sdata[0];
}

// ---------------------------------------------------------------------------
// Tiny MLP head + softmax on the single pooled vector (serial, negligible).
// ---------------------------------------------------------------------------
__global__ void gat_head(const float* __restrict__ GSUM,
                         const float* __restrict__ Wh1, const float* __restrict__ bh1,
                         const float* __restrict__ Wh2, const float* __restrict__ bh2,
                         float* __restrict__ out) {
    if (threadIdx.x != 0 || blockIdx.x != 0) return;
    float g[DIM_H], g1[DIM_H], logit[10];
    for (int d = 0; d < DIM_H; ++d) g[d] = GSUM[d] * (1.0f / (float)N_NODES);
    for (int j = 0; j < DIM_H; ++j) {
        float s = bh1[j];
        for (int d = 0; d < DIM_H; ++d) s += g[d] * Wh1[d * DIM_H + j];
        g1[j] = s > 0.0f ? s : 0.0f;
    }
    float mx = -1e30f;
    for (int c = 0; c < 10; ++c) {
        float s = bh2[c];
        for (int j = 0; j < DIM_H; ++j) s += g1[j] * Wh2[j * 10 + c];
        logit[c] = s;
        if (s > mx) mx = s;
    }
    float den = 0.0f;
    for (int c = 0; c < 10; ++c) { logit[c] = expf(logit[c] - mx); den += logit[c]; }
    for (int c = 0; c < 10; ++c) out[c] = logit[c] / den;
}

// ---------------------------------------------------------------------------
extern "C" void kernel_launch(void* const* d_in, const int* in_sizes, int n_in,
                              void* d_out, int out_size, void* d_ws, size_t ws_size,
                              hipStream_t stream) {
    const float* g_feats = (const float*)d_in[0];
    const int*   e_src   = (const int*)d_in[1];
    const int*   e_dst   = (const int*)d_in[2];
    const float* W_in    = (const float*)d_in[3];
    const float* b_in    = (const float*)d_in[4];
    const float* W1_src  = (const float*)d_in[5];
    const float* b1_src  = (const float*)d_in[6];
    const float* W1_dst  = (const float*)d_in[7];
    const float* b1_dst  = (const float*)d_in[8];
    const float* attn1   = (const float*)d_in[9];
    const float* W2_src  = (const float*)d_in[10];
    const float* b2_src  = (const float*)d_in[11];
    const float* W2_dst  = (const float*)d_in[12];
    const float* b2_dst  = (const float*)d_in[13];
    const float* attn2   = (const float*)d_in[14];
    const float* Wh1     = (const float*)d_in[15];
    const float* bh1     = (const float*)d_in[16];
    const float* Wh2     = (const float*)d_in[17];
    const float* bh2     = (const float*)d_in[18];
    float* out = (float*)d_out;

    // workspace carve-up (floats)
    float* ws = (float*)d_ws;
    float* X    = ws;                              // N*32   (embed out; reused as H)
    float* FS   = X    + (size_t)N_NODES * DIM_H;  // N*128
    float* FD   = FS   + (size_t)N_NODES * HD;     // N*128
    float* OUT  = FD   + (size_t)N_NODES * HD;     // N*128
    float* S    = OUT  + (size_t)N_NODES * HD;     // E*4
    float* SMAX = S    + (size_t)N_EDGES * HEADS;  // N*4
    float* DEN  = SMAX + (size_t)N_NODES * HEADS;  // N*4
    float* GSUM = DEN  + (size_t)N_NODES * HEADS;  // 32

    const int rowTiles = N_NODES / 16;                         // 3125
    const int gemmBlocks = (rowTiles + 7) / 8;                 // 8 waves/block
    const int edgeWaveBlocks = (N_EDGES * 32 + 255) / 256;     // wave per edge
    const int ehBlocks = (N_EDGES * HEADS + 255) / 256;
    const int nhBlocks = (N_NODES * HEADS + 255) / 256;
    const int outBlocks = (N_NODES * HD + 255) / 256;
    const int ndBlocks = (N_NODES * DIM_H + 255) / 256;

    // --- input embedding: [N,64] @ [64,32] ---
    gat_gemm_wmma<DIM_IN, DIM_H><<<gemmBlocks, 256, 0, stream>>>(
        g_feats, W_in, b_in, X, N_NODES);

    // --- GATv2 layer 1 ---
    gat_gemm_wmma<DIM_H, HD><<<gemmBlocks, 256, 0, stream>>>(
        X, W1_src, b1_src, FS, N_NODES);
    gat_gemm_wmma<DIM_H, HD><<<gemmBlocks, 256, 0, stream>>>(
        X, W1_dst, b1_dst, FD, N_NODES);
    gat_fill<<<nhBlocks, 256, 0, stream>>>(SMAX, -1e30f, N_NODES * HEADS);
    gat_fill<<<nhBlocks, 256, 0, stream>>>(DEN, 0.0f, N_NODES * HEADS);
    gat_fill<<<outBlocks, 256, 0, stream>>>(OUT, 0.0f, N_NODES * HD);
    gat_scores<<<edgeWaveBlocks, 256, 0, stream>>>(FS, FD, e_src, e_dst, attn1, S, SMAX);
    gat_expden<<<ehBlocks, 256, 0, stream>>>(e_dst, S, SMAX, DEN);
    gat_aggregate<<<edgeWaveBlocks, 256, 0, stream>>>(FS, e_src, e_dst, S, DEN, OUT);
    gat_headmean<<<ndBlocks, 256, 0, stream>>>(OUT, X, /*relu=*/1);  // X now holds H

    // --- GATv2 layer 2 ---
    gat_gemm_wmma<DIM_H, HD><<<gemmBlocks, 256, 0, stream>>>(
        X, W2_src, b2_src, FS, N_NODES);
    gat_gemm_wmma<DIM_H, HD><<<gemmBlocks, 256, 0, stream>>>(
        X, W2_dst, b2_dst, FD, N_NODES);
    gat_fill<<<nhBlocks, 256, 0, stream>>>(SMAX, -1e30f, N_NODES * HEADS);
    gat_fill<<<nhBlocks, 256, 0, stream>>>(DEN, 0.0f, N_NODES * HEADS);
    gat_fill<<<outBlocks, 256, 0, stream>>>(OUT, 0.0f, N_NODES * HD);
    gat_scores<<<edgeWaveBlocks, 256, 0, stream>>>(FS, FD, e_src, e_dst, attn2, S, SMAX);
    gat_expden<<<ehBlocks, 256, 0, stream>>>(e_dst, S, SMAX, DEN);
    gat_aggregate<<<edgeWaveBlocks, 256, 0, stream>>>(FS, e_src, e_dst, S, DEN, OUT);

    // --- graph readout + MLP head + softmax ---
    gat_colsum<<<32, 256, 0, stream>>>(OUT, GSUM);
    gat_head<<<1, 32, 0, stream>>>(GSUM, Wh1, bh1, Wh2, bh2, out);
}
